// PerformerMultiHeadAttention_79860621902514
// MI455X (gfx1250) — matrix operand order repaired and checked by reading, not compiled, not run on hardware
//
#include <hip/hip_runtime.h>

// ---------------- constants ----------------
#define L_SEQ 8192
#define DM    512
#define NH    8
#define DK    64
#define MF    256        // num random features
#define DEXT  80         // 64 v-dims + 1 ones-column + 15 zero pad (5 WMMA col tiles)

__device__ __constant__ const float kS1       = 0.35355339059327379f; // 64^-0.25
__device__ __constant__ const float kInvSqrtM = 0.0625f;              // 256^-0.5
__device__ __constant__ const float kEps      = 1e-6f;

typedef __attribute__((ext_vector_type(16))) __bf16 bf16x16;
typedef __attribute__((ext_vector_type(8)))  __bf16 bf16x8;
typedef __attribute__((ext_vector_type(8)))  float  floatx8;

// ---------------- WMMA fragment loaders (CDNA5 wave32 layouts) ----------------
// A 16x32 bf16: lanes 0-15 rows 0-15 / half hf: elems 0..7 = K 8*hf..8*hf+7,
//               elems 8..15 = K 16+8*hf..23+8*hf.   "row" points at K=0 of the A-row.
__device__ __forceinline__ bf16x16 frag_a(const __bf16* row, int hf) {
  bf16x8 lo = *(const bf16x8*)(row + 8 * hf);
  bf16x8 hi = *(const bf16x8*)(row + 16 + 8 * hf);
  bf16x16 f;
#pragma unroll
  for (int i = 0; i < 8; ++i) { f[i] = lo[i]; f[8 + i] = hi[i]; }
  return f;
}
// B 32x16 bf16 stored K-contiguous per output column n ("row" points at K=0 of column n):
// lane col = lane&15, half hf: elems 0..7 = K 8*hf..8*hf+7, elems 8..15 = K 16+8*hf..23+8*hf.
__device__ __forceinline__ bf16x16 frag_b(const __bf16* row, int hf) {
  bf16x8 lo = *(const bf16x8*)(row + 8 * hf);
  bf16x8 hi = *(const bf16x8*)(row + 16 + 8 * hf);
  bf16x16 f;
#pragma unroll
  for (int i = 0; i < 8; ++i) { f[i] = lo[i]; f[8 + i] = hi[i]; }
  return f;
}

__device__ __forceinline__ floatx8 wmma_bf16(bf16x16 a, bf16x16 b, floatx8 c) {
  return __builtin_amdgcn_wmma_f32_16x16x32_bf16(false, a, false, b, (short)0, c,
                                                 false, false);
}

__device__ __forceinline__ floatx8 zero8() {
  floatx8 z = {0.f, 0.f, 0.f, 0.f, 0.f, 0.f, 0.f, 0.f};
  return z;
}

// ---------------- Kernel 1/6: C[32768,512] = A[32768,512] * W[512,512] ----------------
// Double-buffered LDS; global loads for tile i+1 issued before WMMA of tile i.
// splitHeads=1 : write bf16 into [B,H,L,64] head-split layout (Q/K/V projections)
// splitHeads=0 : write f32 row-major (final Wo projection -> d_out)
__global__ __launch_bounds__(256) void gemm512_kernel(
    const float* __restrict__ A, const float* __restrict__ W,
    __bf16* __restrict__ outHeads, float* __restrict__ outFlat, int splitHeads) {
  __shared__ __bf16 sA[2][128 * 40];  // [row][k] K-contiguous, pad to 40
  __shared__ __bf16 sB[2][64 * 40];   // [n][k]   K-contiguous, pad to 40
  const int t = threadIdx.x, lane = t & 31, w = t >> 5;
  const int ln = lane & 15, hf = lane >> 4;
  const int r0 = blockIdx.x * 128, n0 = blockIdx.y * 64;
  const int mw = w >> 1, nw = w & 1;  // 4x2 wave grid, 32x32 per wave

  float4 ra[4], rb[2];
  auto gload = [&](int k0) {
#pragma unroll
    for (int i = 0; i < 4; ++i) {            // A tile 128x32, 8 float4 per row
      int c4 = t + i * 256;
      int row = c4 >> 3, kk = (c4 & 7) * 4;
      ra[i] = *(const float4*)(A + (size_t)(r0 + row) * 512 + k0 + kk);
    }
#pragma unroll
    for (int i = 0; i < 2; ++i) {            // B tile 32x64, 16 float4 per k-row
      int c4 = t + i * 256;
      int kk = c4 >> 4, nn = (c4 & 15) * 4;
      rb[i] = *(const float4*)(W + (size_t)(k0 + kk) * 512 + n0 + nn);
    }
  };
  auto lstore = [&](int p) {
#pragma unroll
    for (int i = 0; i < 4; ++i) {
      int c4 = t + i * 256;
      int row = c4 >> 3, kk = (c4 & 7) * 4;
      __bf16* d = sA[p] + row * 40 + kk;
      d[0] = (__bf16)ra[i].x; d[1] = (__bf16)ra[i].y;
      d[2] = (__bf16)ra[i].z; d[3] = (__bf16)ra[i].w;
    }
#pragma unroll
    for (int i = 0; i < 2; ++i) {            // transposed: [n][k]
      int c4 = t + i * 256;
      int kk = c4 >> 4, nn = (c4 & 15) * 4;
      sB[p][(nn + 0) * 40 + kk] = (__bf16)rb[i].x;
      sB[p][(nn + 1) * 40 + kk] = (__bf16)rb[i].y;
      sB[p][(nn + 2) * 40 + kk] = (__bf16)rb[i].z;
      sB[p][(nn + 3) * 40 + kk] = (__bf16)rb[i].w;
    }
  };

  floatx8 acc[2][2];
#pragma unroll
  for (int i = 0; i < 2; ++i)
#pragma unroll
    for (int j = 0; j < 2; ++j) acc[i][j] = zero8();

  gload(0);
  lstore(0);
  for (int it = 0; it < 16; ++it) {
    __syncthreads();
    if (it + 1 < 16) gload((it + 1) * 32);   // next tile in flight during WMMA
    if (it + 2 < 16)                          // gfx1250 global_prefetch two ahead
      __builtin_prefetch(A + (size_t)(r0 + (t >> 1)) * 512 + (it + 2) * 32, 0, 1);
    const int cur = it & 1;
    bf16x16 af[2], bfv[2];
#pragma unroll
    for (int i = 0; i < 2; ++i)
      af[i] = frag_a(sA[cur] + (mw * 32 + i * 16 + ln) * 40, hf);
#pragma unroll
    for (int j = 0; j < 2; ++j)
      bfv[j] = frag_b(sB[cur] + (nw * 32 + j * 16 + ln) * 40, hf);
#pragma unroll
    for (int i = 0; i < 2; ++i)
#pragma unroll
      for (int j = 0; j < 2; ++j) acc[i][j] = wmma_bf16(af[i], bfv[j], acc[i][j]);
    if (it + 1 < 16) lstore((it + 1) & 1);   // other buffer: no barrier needed here
  }

  // epilogue: C tile layout -> row = e + 8*hf, col = ln
#pragma unroll
  for (int i = 0; i < 2; ++i)
#pragma unroll
    for (int j = 0; j < 2; ++j)
#pragma unroll
      for (int e = 0; e < 8; ++e) {
        int r = r0 + mw * 32 + i * 16 + e + 8 * hf;
        int n = n0 + nw * 32 + j * 16 + ln;
        float v = acc[i][j][e];
        if (splitHeads) {
          int b = r >> 13, l = r & (L_SEQ - 1);
          int h = n >> 6, d = n & 63;
          outHeads[(((size_t)b * NH + h) * L_SEQ + l) * DK + d] = (__bf16)v;
        } else {
          outFlat[(size_t)r * DM + n] = v;
        }
      }
}

// ---------------- Kernel 2: FAVOR+ features  P = exp(X*s1 @ omega^T - sq - stab)/16 + eps
__global__ __launch_bounds__(256) void phi_kernel(const __bf16* __restrict__ X,
                                                  const float* __restrict__ omega,
                                                  __bf16* __restrict__ P) {
  __shared__ __bf16 sX[128 * 72];   // [row][k=64]
  __shared__ __bf16 sO[256 * 72];   // [feature n][k=64], pre-scaled by s1
  __shared__ float sSq[128];
  const int t = threadIdx.x, lane = t & 31, w = t >> 5;
  const int ln = lane & 15, hf = lane >> 4;
  const int l0 = blockIdx.x * 128;
  const size_t bh = blockIdx.y;

  // batch all global loads first, then stage into LDS
  float4 ro[16];
  bf16x8 rx[4];
#pragma unroll
  for (int i = 0; i < 16; ++i) {
    int c4 = t + i * 256;                  // 4096 float4 chunks of omega
    int n = c4 >> 4, k = (c4 & 15) * 4;
    ro[i] = *(const float4*)(omega + (size_t)n * DK + k);
  }
#pragma unroll
  for (int i = 0; i < 4; ++i) {
    int c8 = t + i * 256;                  // 1024 chunks of 8 bf16 of X
    int row = c8 >> 3, k = (c8 & 7) * 8;
    rx[i] = *(const bf16x8*)(X + (bh * L_SEQ + l0 + row) * (size_t)DK + k);
  }
#pragma unroll
  for (int i = 0; i < 16; ++i) {
    int c4 = t + i * 256;
    int n = c4 >> 4, k = (c4 & 15) * 4;
    __bf16* d = sO + n * 72 + k;
    d[0] = (__bf16)(ro[i].x * kS1); d[1] = (__bf16)(ro[i].y * kS1);
    d[2] = (__bf16)(ro[i].z * kS1); d[3] = (__bf16)(ro[i].w * kS1);
  }
#pragma unroll
  for (int i = 0; i < 4; ++i) {
    int c8 = t + i * 256;
    int row = c8 >> 3, k = (c8 & 7) * 8;
    *(bf16x8*)(sX + row * 72 + k) = rx[i];
  }
  __syncthreads();
  if (t < 128) {
    const __bf16* xr = sX + t * 72;
    float s = 0.f;
#pragma unroll
    for (int d = 0; d < DK; ++d) { float x = (float)xr[d] * kS1; s += x * x; }
    sSq[t] = 0.5f * s;
  }
  __syncthreads();

  // one wave owns 16 rows x all 256 features
  bf16x16 a0 = frag_a(sX + (w * 16 + ln) * 72, hf);
  bf16x16 a1 = frag_a(sX + (w * 16 + ln) * 72 + 32, hf);
  floatx8 acc[16];
#pragma unroll
  for (int c = 0; c < 16; ++c) acc[c] = zero8();
#pragma unroll
  for (int c = 0; c < 16; ++c) {
    bf16x16 b0 = frag_b(sO + (c * 16 + ln) * 72, hf);
    bf16x16 b1 = frag_b(sO + (c * 16 + ln) * 72 + 32, hf);
    acc[c] = wmma_bf16(a0, b0, acc[c]);
    acc[c] = wmma_bf16(a1, b1, acc[c]);
  }

  // per-row max over 256 features: partial over col tiles, then 16-lane xor tree
  float stab[8];
#pragma unroll
  for (int e = 0; e < 8; ++e) {
    float m = acc[0][e];
#pragma unroll
    for (int c = 1; c < 16; ++c) m = fmaxf(m, acc[c][e]);
    m = fmaxf(m, __shfl_xor(m, 1, 32));
    m = fmaxf(m, __shfl_xor(m, 2, 32));
    m = fmaxf(m, __shfl_xor(m, 4, 32));
    m = fmaxf(m, __shfl_xor(m, 8, 32));
    stab[e] = m;
  }
#pragma unroll
  for (int e = 0; e < 8; ++e) {
    int rloc = w * 16 + e + 8 * hf;
    float sq = sSq[rloc];
    size_t rowAddr = (bh * L_SEQ + l0 + rloc) * (size_t)MF;
#pragma unroll
    for (int c = 0; c < 16; ++c) {
      float v = __expf(acc[c][e] - sq - stab[e]) * kInvSqrtM + kEps;
      P[rowAddr + c * 16 + ln] = (__bf16)v;
    }
  }
}

// ---------------- Kernel 3: kv partial: C[256,80] += kp^T chunk @ [v|1|0] ----------------
// Double-buffered; constant ones/zero columns written once for both buffers.
__global__ __launch_bounds__(256) void kv_partial_kernel(
    const __bf16* __restrict__ KP, const __bf16* __restrict__ VH,
    float* __restrict__ part) {
  __shared__ __bf16 sA[2][256 * 40];  // [m][l] (transposed kp)
  __shared__ __bf16 sB[2][80 * 40];   // [d][l]
  const int t = threadIdx.x, lane = t & 31, w = t >> 5;
  const int ln = lane & 15, hf = lane >> 4;
  const int kc = blockIdx.x;       // 16 L-chunks of 512
  const size_t bh = blockIdx.y;
  const int li = t >> 3, ms = (t & 7) * 32, d0 = (t & 7) * 8;

  bf16x8 rA[4], rV;
  auto gload = [&](int it) {
    int l0 = kc * 512 + it * 32;
    const __bf16* src = KP + (bh * L_SEQ + l0 + li) * (size_t)MF + ms;
#pragma unroll
    for (int j = 0; j < 4; ++j) rA[j] = *(const bf16x8*)(src + j * 8);
    rV = *(const bf16x8*)(VH + (bh * L_SEQ + l0 + li) * (size_t)DK + d0);
  };
  auto lstore = [&](int p) {
#pragma unroll
    for (int j = 0; j < 4; ++j)
#pragma unroll
      for (int e = 0; e < 8; ++e) sA[p][(ms + j * 8 + e) * 40 + li] = rA[j][e];
#pragma unroll
    for (int e = 0; e < 8; ++e) sB[p][(d0 + e) * 40 + li] = rV[e];
  };

  // ones column (d=64) + zero pad, once, both buffers
  for (int idx = t; idx < 1024; idx += 256) {
    int p = idx >> 9, r = idx & 511;
    int d = 64 + (r >> 5), l = r & 31;
    sB[p][d * 40 + l] = (d == 64) ? (__bf16)1.0f : (__bf16)0.0f;
  }

  floatx8 acc[2][5];
#pragma unroll
  for (int i = 0; i < 2; ++i)
#pragma unroll
    for (int j = 0; j < 5; ++j) acc[i][j] = zero8();

  gload(0);
  lstore(0);
  for (int it = 0; it < 16; ++it) {
    __syncthreads();
    if (it + 1 < 16) gload(it + 1);
    const int cur = it & 1;
    bf16x16 af[2], bfv[5];
#pragma unroll
    for (int i = 0; i < 2; ++i)
      af[i] = frag_a(sA[cur] + (w * 32 + i * 16 + ln) * 40, hf);
#pragma unroll
    for (int j = 0; j < 5; ++j)
      bfv[j] = frag_b(sB[cur] + (j * 16 + ln) * 40, hf);
#pragma unroll
    for (int i = 0; i < 2; ++i)
#pragma unroll
      for (int j = 0; j < 5; ++j) acc[i][j] = wmma_bf16(af[i], bfv[j], acc[i][j]);
    if (it + 1 < 16) lstore((it + 1) & 1);
  }
#pragma unroll
  for (int i = 0; i < 2; ++i)
#pragma unroll
    for (int j = 0; j < 5; ++j)
#pragma unroll
      for (int e = 0; e < 8; ++e) {
        int m = w * 32 + i * 16 + e + 8 * hf;
        int n = j * 16 + ln;
        part[((bh * 16 + kc) * 256 + m) * (size_t)DEXT + n] = acc[i][j][e];
      }
}

// ---------------- Kernel 4: reduce 16 partials, emit bf16 kv ----------------
__global__ __launch_bounds__(256) void kv_reduce_kernel(const float* __restrict__ part,
                                                        __bf16* __restrict__ kvb) {
  int idx = blockIdx.x * 256 + threadIdx.x;  // 32*256*80 = 655360
  if (idx >= 32 * 256 * DEXT) return;
  int bh = idx / (256 * DEXT);
  int rem = idx - bh * (256 * DEXT);
  float s = 0.f;
#pragma unroll
  for (int c = 0; c < 16; ++c)
    s += part[((size_t)bh * 16 + c) * (256 * DEXT) + rem];
  kvb[idx] = (__bf16)s;
}

// ---------------- Kernel 5: num = qp @ kv_ext, normalize by den (col 64) ----------------
__global__ __launch_bounds__(256) void num_norm_kernel(const __bf16* __restrict__ QP,
                                                       const __bf16* __restrict__ KVB,
                                                       float* __restrict__ outPre) {
  __shared__ __bf16 sA[128 * 72];   // qp tile, 64 K at a time
  __shared__ __bf16 sB[80 * 264];   // [n=80][k=256]
  const int t = threadIdx.x, lane = t & 31, w = t >> 5;
  const int ln = lane & 15, hf = lane >> 4;
  const int l0 = blockIdx.x * 128;
  const int bh = blockIdx.y, b = bh >> 3, h = bh & 7;

  for (int idx = t; idx < 256 * DEXT; idx += 256) {  // kv[m][n] -> sB[n][m]
    int m = idx / DEXT, n = idx - m * DEXT;
    sB[n * 264 + m] = KVB[((size_t)bh * 256 + m) * DEXT + n];
  }
  floatx8 acc[5];
#pragma unroll
  for (int c = 0; c < 5; ++c) acc[c] = zero8();

  for (int kk = 0; kk < 256; kk += 64) {
    __syncthreads();  // first iter: sB ready; later: sA reuse safe
    bf16x8 rq[4];
#pragma unroll
    for (int i = 0; i < 4; ++i) {
      int c8 = t + i * 256;
      int row = c8 >> 3, k = (c8 & 7) * 8;
      rq[i] = *(const bf16x8*)(QP + ((size_t)bh * L_SEQ + l0 + row) * MF + kk + k);
    }
#pragma unroll
    for (int i = 0; i < 4; ++i) {
      int c8 = t + i * 256;
      int row = c8 >> 3, k = (c8 & 7) * 8;
      *(bf16x8*)(sA + row * 72 + k) = rq[i];
    }
    __syncthreads();
#pragma unroll
    for (int inner = 0; inner < 2; ++inner) {
      bf16x16 a = frag_a(sA + (w * 16 + ln) * 72 + inner * 32, hf);
#pragma unroll
      for (int c = 0; c < 5; ++c) {
        bf16x16 bb = frag_b(sB + (c * 16 + ln) * 264 + kk + inner * 32, hf);
        acc[c] = wmma_bf16(a, bb, acc[c]);
      }
    }
  }
  // den lives at n=64 (col tile 4, ln==0); broadcast within half-wave
#pragma unroll
  for (int e = 0; e < 8; ++e) {
    float den = __shfl(acc[4][e], lane & 16, 32);
    float inv = 1.0f / (den + kEps);
    int l = l0 + w * 16 + e + 8 * hf;
    size_t base = ((size_t)b * L_SEQ + l) * DM + h * DK;
#pragma unroll
    for (int c = 0; c < 4; ++c) outPre[base + c * 16 + ln] = acc[c][e] * inv;
  }
}

// ---------------- host-side launcher ----------------
extern "C" void kernel_launch(void* const* d_in, const int* in_sizes, int n_in,
                              void* d_out, int out_size, void* d_ws, size_t ws_size,
                              hipStream_t stream) {
  const float* Q  = (const float*)d_in[0];
  const float* K  = (const float*)d_in[1];
  const float* V  = (const float*)d_in[2];
  const float* Wq = (const float*)d_in[3];
  const float* Wk = (const float*)d_in[4];
  const float* Wv = (const float*)d_in[5];
  const float* Wo = (const float*)d_in[6];
  const float* omega = (const float*)d_in[7];
  float* out = (float*)d_out;

  const size_t NHLD = (size_t)4 * NH * L_SEQ * DK;   // head-split q/k/v elems
  const size_t NP   = (size_t)4 * NH * L_SEQ * MF;   // qp/kp elems
  char* ws = (char*)d_ws;
  size_t off = 0;
  auto take = [&](size_t bytes) { size_t o = off; off = (off + bytes + 255) & ~(size_t)255; return o; };
  __bf16* qh   = (__bf16*)(ws + take(NHLD * 2));
  __bf16* kh   = (__bf16*)(ws + take(NHLD * 2));
  __bf16* vh   = (__bf16*)(ws + take(NHLD * 2));
  __bf16* qp   = (__bf16*)(ws + take(NP * 2));
  __bf16* kp   = (__bf16*)(ws + take(NP * 2));
  float*  part = (float*)(ws + take((size_t)32 * 16 * 256 * DEXT * 4));
  __bf16* kvb  = (__bf16*)(ws + take((size_t)32 * 256 * DEXT * 2));
  float*  pre  = (float*)(ws + take((size_t)4 * L_SEQ * DM * 4));

  dim3 blk(256);
  dim3 gGemm(256, 8);    // 32768/128 x 512/64
  dim3 gPhi(64, 32);     // 8192/128 x B*H
  dim3 gKv(16, 32);      // 16 L-chunks x B*H
  dim3 gNum(64, 32);

  gemm512_kernel<<<gGemm, blk, 0, stream>>>(Q, Wq, qh, nullptr, 1);
  gemm512_kernel<<<gGemm, blk, 0, stream>>>(K, Wk, kh, nullptr, 1);
  gemm512_kernel<<<gGemm, blk, 0, stream>>>(V, Wv, vh, nullptr, 1);
  phi_kernel<<<gPhi, blk, 0, stream>>>(qh, omega, qp);
  phi_kernel<<<gPhi, blk, 0, stream>>>(kh, omega, kp);
  kv_partial_kernel<<<gKv, blk, 0, stream>>>(kp, vh, part);
  kv_reduce_kernel<<<2560, blk, 0, stream>>>(part, kvb);
  num_norm_kernel<<<gNum, blk, 0, stream>>>(qp, kvb, pre);
  gemm512_kernel<<<gGemm, blk, 0, stream>>>(pre, Wo, nullptr, out, 0);
}
